// MultiheadedFeatureSelfAttention_11802570130120
// MI455X (gfx1250) — compile-verified, matrix operand order checked by reading, not run on hardware
//
#include <hip/hip_runtime.h>
#include <hip/hip_bf16.h>
#include <math.h>

// ---------------------------------------------------------------------------
// MultiheadedFeatureSelfAttention for MI455X (gfx1250), wave32 + WMMA bf16
// + Tensor Data Mover (TDM) double-buffered K/V staging in the attention core.
//   q = GELU(query@Wq1+bq1)@Wq2+bq2      [B,S,256]  (bf16 intermediate)
//   k = GELU(key  @Wk1+bk1)@Wk2+bk2      [B,S,256]  (bf16)
//   v = value@Wv+bv                      [B,S,512]  (bf16)
//   x = flash_attention(q,k,v)           [B,S,512]  (bf16)
//   out = x@Wo+bo                        [B,S,512]  (fp32)
// ---------------------------------------------------------------------------

typedef __attribute__((ext_vector_type(16))) __bf16 v16bf;
typedef __attribute__((ext_vector_type(8)))  float  v8f;
typedef __attribute__((ext_vector_type(4)))  unsigned int v4ui;
typedef __attribute__((ext_vector_type(8)))  int    v8si;
typedef __attribute__((ext_vector_type(4)))  int    v4si;

static constexpr int BATCH  = 2;
static constexpr int SEQ    = 4096;
static constexpr int DMODEL = 512;
static constexpr int HEADS  = 8;
static constexpr int DK     = 64;   // per-head value dim
static constexpr int DQK    = 32;   // per-head q/k dim
static constexpr int MROWS  = BATCH * SEQ;  // 8192

// ---- LDS byte offset of a __shared__ object (generic -> AS3 -> u32) --------
typedef const char __attribute__((address_space(3))) lds_cchar;
__device__ __forceinline__ unsigned lds_off(const void* p) {
  return (unsigned)(unsigned long long)(lds_cchar*)p;
}

// ---- TDM: 2-D tile load, bf16 (data_size=2B), LDS row padding --------------
// D# layout per CDNA5 ISA 8.3-8.6. group0: count=1, lds_addr, global_addr,
// type=2. group1: data_size=1(2B), pad ctl, tensor dims/strides, tile dims.
// This toolchain exposes the 6-arg builtin: (g0, g1, g2, g3, g_ext, cpol).
__device__ __forceinline__ void tdm_load_2d_bf16(
    const void* gsrc, unsigned ldsoff,
    unsigned tensor_d0, unsigned tensor_d1, unsigned stride0_elems,
    unsigned tile_d0, unsigned tile_d1,
    unsigned pad_interval_code, unsigned pad_amount_code) {
  unsigned long long ga = (unsigned long long)gsrc;
  v4ui g0;
  g0.x = 1u;                                         // count=1 (valid), user mode
  g0.y = ldsoff;                                     // lds_addr (bytes)
  g0.z = (unsigned)ga;                               // global_addr[31:0]
  g0.w = ((unsigned)(ga >> 32) & 0x01ffffffu) | (2u << 30);  // [56:32] | type=2
  v8si g1;
  unsigned w0 = (1u << 16)                           // data_size = 2 bytes
              | (1u << 20)                           // pad_enable
              | (pad_interval_code << 22)
              | (pad_amount_code << 25);
  g1[0] = (int)w0;                                   // wg_mask=0 (no cluster)
  g1[1] = (int)((tensor_d0 & 0xffffu) << 16);        // tensor_dim0[15:0]
  g1[2] = (int)((tensor_d0 >> 16) | ((tensor_d1 & 0xffffu) << 16));
  g1[3] = (int)((tensor_d1 >> 16) | (tile_d0 << 16));// tile_dim0
  g1[4] = (int)(tile_d1 & 0xffffu);                  // tile_dim1, tile_dim2=0
  g1[5] = (int)stride0_elems;                        // tensor_dim0_stride[31:0]
  g1[6] = 0;                                         // stride hi / dim1_stride
  g1[7] = 0;
  v4si g2 = {0, 0, 0, 0};                            // 2-D: groups 2/3 unused
  v4si g3 = {0, 0, 0, 0};
  v8si gx = {0, 0, 0, 0, 0, 0, 0, 0};                // extension group (unused)
  __builtin_amdgcn_tensor_load_to_lds(g0, g1, g2, g3, gx, 0);
}

// ---- WMMA fragment loaders (ISA 7.12.2, 16-bit 16x32 A layout) -------------
// A (16x32, MxK): lane = m%16 + 16*half; vgpr pair j:
//   kk = ((j&4)<<2) + half*8 + (j&3)*2   -> {kk, kk+1}
__device__ __forceinline__ v16bf frag_a_rowmajor(const __bf16* base, int stride,
                                                 int m_local, int half) {
  v16bf f;
  const __bf16* row = base + m_local * stride;
#pragma unroll
  for (int j = 0; j < 8; ++j) {
    int kk = ((j & 4) << 2) + half * 8 + (j & 3) * 2;
    f[2 * j]     = row[kk];
    f[2 * j + 1] = row[kk + 1];
  }
  return f;
}

// B (32x16, KxN) stored [k][n]: lane = n%16 + 16*half; same kk mapping.
__device__ __forceinline__ v16bf frag_b_kmajor(const __bf16* base, int stride,
                                               int n_local, int half) {
  v16bf f;
#pragma unroll
  for (int j = 0; j < 8; ++j) {
    int kk = ((j & 4) << 2) + half * 8 + (j & 3) * 2;
    f[2 * j]     = base[kk * stride + n_local];
    f[2 * j + 1] = base[(kk + 1) * stride + n_local];
  }
  return f;
}

// ---------------------------------------------------------------------------
// GEMM:  C[M,N] = act(A[M,K] @ W[K,N] + bias[N]),  M%128==0, N%128==0, K%32==0
// 256 threads = 8 waves; workgroup tile 128x128; wave tile 32x64 (2x4 WMMA).
// ---------------------------------------------------------------------------
template <bool GELU_ACT, typename InT, typename OutT>
__global__ __launch_bounds__(256) void gemm_bias_act(
    const InT* __restrict__ A, const float* __restrict__ W,
    const float* __restrict__ bias, OutT* __restrict__ C,
    int M, int N, int K) {
  __shared__ __bf16 As[128][40];   // [m][k] + pad
  __shared__ __bf16 Bs[32][136];   // [k][n] + pad

  const int tid  = threadIdx.x;
  const int wave = tid >> 5, lane = tid & 31;
  const int wm = wave >> 1, wn = wave & 1;     // 4x2 wave grid
  const int m0 = blockIdx.y * 128;
  const int n0 = blockIdx.x * 128;
  const int loc  = lane & 15;
  const int half = lane >> 4;

  v8f acc[2][4] = {};

  for (int k0 = 0; k0 < K; k0 += 32) {
#pragma unroll
    for (int t = 0; t < 16; ++t) {             // A tile 128x32 -> bf16
      int idx = tid + t * 256;
      int r = idx >> 5, c = idx & 31;
      As[r][c] = (__bf16)A[(size_t)(m0 + r) * K + (k0 + c)];
    }
#pragma unroll
    for (int t = 0; t < 16; ++t) {             // W tile 32x128 -> bf16
      int idx = tid + t * 256;
      int r = idx >> 7, c = idx & 127;
      Bs[r][c] = (__bf16)W[(size_t)(k0 + r) * N + (n0 + c)];
    }
    __syncthreads();

    v16bf af[2], bf[4];
#pragma unroll
    for (int i = 0; i < 2; ++i)
      af[i] = frag_a_rowmajor(&As[wm * 32 + i * 16][0], 40, loc, half);
#pragma unroll
    for (int j = 0; j < 4; ++j)
      bf[j] = frag_b_kmajor(&Bs[0][wn * 64 + j * 16], 136, loc, half);

#pragma unroll
    for (int i = 0; i < 2; ++i)
#pragma unroll
      for (int j = 0; j < 4; ++j)
        acc[i][j] = __builtin_amdgcn_wmma_f32_16x16x32_bf16(
            false, af[i], false, bf[j], (short)0, acc[i][j], false, false);

    __syncthreads();
  }

#pragma unroll
  for (int i = 0; i < 2; ++i) {
#pragma unroll
    for (int j = 0; j < 4; ++j) {
#pragma unroll
      for (int r = 0; r < 8; ++r) {
        int row = m0 + wm * 32 + i * 16 + r + 8 * half;
        int col = n0 + wn * 64 + j * 16 + loc;
        float v = acc[i][j][r] + bias[col];
        if (GELU_ACT)
          v = 0.5f * v * (1.0f + erff(v * 0.7071067811865476f));  // exact GELU
        C[(size_t)row * N + col] = (OutT)v;
      }
    }
  }
}

// ---------------------------------------------------------------------------
// Flash attention: one workgroup = 128 query rows of one (b,h).
// Each wave owns 16 q rows; TDM double-buffers 32-key K/V tiles into LDS.
// q,k: [B,S,H*DQK] bf16 ; v,x: [B,S,H*DK] bf16
// ---------------------------------------------------------------------------
__global__ __launch_bounds__(256) void flash_attn(
    const __bf16* __restrict__ Qg, const __bf16* __restrict__ Kg,
    const __bf16* __restrict__ Vg, __bf16* __restrict__ Xg) {
  __shared__ __bf16 Ks[2][32][40];    // [buf][key][dim], TDM pads 64B->80B rows
  __shared__ __bf16 Vs[2][32][72];    // [buf][key][dim], TDM pads 128B->144B
  __shared__ __bf16 Ps[8][16][40];    // per-wave P tile [qrow][key]

  const int tid  = threadIdx.x;
  const int wave = tid >> 5, lane = tid & 31;
  const int loc  = lane & 15;
  const int half = lane >> 4;
  const int b = blockIdx.z, h = blockIdx.y;
  const int q0 = blockIdx.x * 128 + wave * 16;
  const float scale = 0.17677669529663688f;  // 1/sqrt(32)
  constexpr int NT = SEQ / 32;               // 128 key tiles

  // Q fragment (A layout) held in registers for the whole key sweep.
  v16bf qf;
  {
    const __bf16* qp = Qg + ((size_t)(b * SEQ + q0 + loc)) * (HEADS * DQK) + h * DQK;
#pragma unroll
    for (int j = 0; j < 8; ++j) {
      int kk = ((j & 4) << 2) + half * 8 + (j & 3) * 2;
      qf[2 * j]     = qp[kk];
      qf[2 * j + 1] = qp[kk + 1];
    }
  }

  v8f o[4] = {};
  float mi[8], li[8];
#pragma unroll
  for (int r = 0; r < 8; ++r) { mi[r] = -3.402823466e38f; li[r] = 0.0f; }

  // TDM tile issue (wave 0 only): K tile 32x32, V tile 32x64, bf16 rows.
  auto issue = [&](int kb, int p) {
    const __bf16* kg = Kg + ((size_t)(b * SEQ + kb)) * (HEADS * DQK) + h * DQK;
    const __bf16* vg = Vg + ((size_t)(b * SEQ + kb)) * (HEADS * DK) + h * DK;
    // K: rows of 32 bf16 (16 dwords) + 4 dwords pad -> stride 40 elems
    tdm_load_2d_bf16(kg, lds_off(&Ks[p][0][0]),
                     HEADS * DQK, SEQ, HEADS * DQK, DQK, 32,
                     /*interval 16 dwords*/ 3, /*pad 4 dwords*/ 3);
    // V: rows of 64 bf16 (32 dwords) + 4 dwords pad -> stride 72 elems
    tdm_load_2d_bf16(vg, lds_off(&Vs[p][0][0]),
                     HEADS * DK, SEQ, HEADS * DK, DK, 32,
                     /*interval 32 dwords*/ 4, /*pad 4 dwords*/ 3);
  };

  if (wave == 0) issue(0, 0);

  for (int t = 0; t < NT; ++t) {
    const int p = t & 1;
    if (wave == 0) {
      if (t + 1 < NT) {
        issue((t + 1) * 32, p ^ 1);
        __builtin_amdgcn_s_wait_tensorcnt(2);  // tile t's 2 loads complete
      } else {
        __builtin_amdgcn_s_wait_tensorcnt(0);
      }
    }
    __syncthreads();

    // scores S = Q @ K^T : B[kk=dim][n=key] == Ks[key][dim] -> A-style indexing
    v16bf kf0 = frag_a_rowmajor(&Ks[p][0][0],  40, loc, half);
    v16bf kf1 = frag_a_rowmajor(&Ks[p][16][0], 40, loc, half);
    v8f z = {};
    v8f s0 = __builtin_amdgcn_wmma_f32_16x16x32_bf16(false, qf, false, kf0,
                                                     (short)0, z, false, false);
    v8f s1 = __builtin_amdgcn_wmma_f32_16x16x32_bf16(false, qf, false, kf1,
                                                     (short)0, z, false, false);
    s0 *= scale;
    s1 *= scale;

    // online softmax per row r (row = r + 8*half, cols spread over 16 lanes)
#pragma unroll
    for (int r = 0; r < 8; ++r) {
      float vmax = fmaxf(s0[r], s1[r]);
#pragma unroll
      for (int off = 8; off >= 1; off >>= 1)
        vmax = fmaxf(vmax, __shfl_xor(vmax, off, 32));
      float mn = fmaxf(mi[r], vmax);
      float al = __expf(mi[r] - mn);
      float p0 = __expf(s0[r] - mn);
      float p1 = __expf(s1[r] - mn);
      float rs = p0 + p1;
#pragma unroll
      for (int off = 8; off >= 1; off >>= 1)
        rs += __shfl_xor(rs, off, 32);
      li[r] = li[r] * al + rs;
      mi[r] = mn;
#pragma unroll
      for (int g = 0; g < 4; ++g) o[g][r] *= al;
      Ps[wave][r + 8 * half][loc]      = (__bf16)p0;   // D layout -> row major
      Ps[wave][r + 8 * half][16 + loc] = (__bf16)p1;
    }
    // wave-local LDS RAW: D-layout stores -> A-layout loads
    asm volatile("s_wait_dscnt 0" ::: "memory");

    v16bf pf = frag_a_rowmajor(&Ps[wave][0][0], 40, loc, half);
#pragma unroll
    for (int g = 0; g < 4; ++g) {
      v16bf vf = frag_b_kmajor(&Vs[p][0][g * 16], 72, loc, half);
      o[g] = __builtin_amdgcn_wmma_f32_16x16x32_bf16(false, pf, false, vf,
                                                     (short)0, o[g], false, false);
    }
    __syncthreads();  // buffer p reusable only after all waves consumed it
  }

  // normalize and write x[b, qrow, h*64 + col] (bf16)
#pragma unroll
  for (int g = 0; g < 4; ++g) {
#pragma unroll
    for (int r = 0; r < 8; ++r) {
      int qrow = q0 + r + 8 * half;
      float val = o[g][r] / li[r];
      Xg[((size_t)(b * SEQ + qrow)) * (HEADS * DK) + h * DK + g * 16 + loc] =
          (__bf16)val;
    }
  }
}

// ---------------------------------------------------------------------------
extern "C" void kernel_launch(void* const* d_in, const int* in_sizes, int n_in,
                              void* d_out, int out_size, void* d_ws, size_t ws_size,
                              hipStream_t stream) {
  const float* query = (const float*)d_in[0];
  const float* key   = (const float*)d_in[1];
  const float* value = (const float*)d_in[2];
  const float* Wq1 = (const float*)d_in[3];
  const float* bq1 = (const float*)d_in[4];
  const float* Wq2 = (const float*)d_in[5];
  const float* bq2 = (const float*)d_in[6];
  const float* Wk1 = (const float*)d_in[7];
  const float* bk1 = (const float*)d_in[8];
  const float* Wk2 = (const float*)d_in[9];
  const float* bk2 = (const float*)d_in[10];
  const float* Wv  = (const float*)d_in[11];
  const float* bv  = (const float*)d_in[12];
  const float* Wo  = (const float*)d_in[13];
  const float* bo  = (const float*)d_in[14];
  float* out = (float*)d_out;

  // workspace (bf16): hidden(16MB) | q(4MB) | k(4MB) | v(8MB) | x(8MB)
  __bf16* ws  = (__bf16*)d_ws;
  __bf16* hid = ws;                                 // 8192 x 1024
  __bf16* qb  = hid + (size_t)MROWS * 1024;         // 8192 x 256
  __bf16* kb  = qb  + (size_t)MROWS * 256;          // 8192 x 256
  __bf16* vb  = kb  + (size_t)MROWS * 256;          // 8192 x 512
  __bf16* xb  = vb  + (size_t)MROWS * 512;          // 8192 x 512

  const dim3 blk(256);

  // q path
  gemm_bias_act<true,  float,  __bf16><<<dim3(1024 / 128, MROWS / 128), blk, 0, stream>>>(
      query, Wq1, bq1, hid, MROWS, 1024, DMODEL);
  gemm_bias_act<false, __bf16, __bf16><<<dim3(256 / 128, MROWS / 128), blk, 0, stream>>>(
      hid, Wq2, bq2, qb, MROWS, 256, 1024);
  // k path (reuses hidden buffer)
  gemm_bias_act<true,  float,  __bf16><<<dim3(1024 / 128, MROWS / 128), blk, 0, stream>>>(
      key, Wk1, bk1, hid, MROWS, 1024, DMODEL);
  gemm_bias_act<false, __bf16, __bf16><<<dim3(256 / 128, MROWS / 128), blk, 0, stream>>>(
      hid, Wk2, bk2, kb, MROWS, 256, 1024);
  // v path
  gemm_bias_act<false, float,  __bf16><<<dim3(512 / 128, MROWS / 128), blk, 0, stream>>>(
      value, Wv, bv, vb, MROWS, 512, DMODEL);
  // attention (TDM-staged K/V)
  flash_attn<<<dim3(SEQ / 128, HEADS, BATCH), blk, 0, stream>>>(qb, kb, vb, xb);
  // output projection (fp32 out)
  gemm_bias_act<false, __bf16, float><<<dim3(512 / 128, MROWS / 128), blk, 0, stream>>>(
      xb, Wo, bo, out, MROWS, 512, DMODEL);
}
